// LinearAttention_8143257993984
// MI455X (gfx1250) — matrix operand order repaired and checked by reading, not compiled, not run on hardware
//
#include <hip/hip_runtime.h>
#include <hip/hip_bf16.h>

#define DMODEL 1024
#define NHEADS 16
#define HDIM   64
#define BATCH  4
#define SEQ    4096
#define NTOK   (BATCH * SEQ)
#define SPLIT  8                       // S-dim split for kv reduction
#define SCHUNK (SEQ / SPLIT)           // 512

typedef __attribute__((ext_vector_type(16))) __bf16 v16bf;
typedef __attribute__((ext_vector_type(8)))  float  v8f;

union AFrag { unsigned int u[8]; v16bf v; };

__device__ __forceinline__ unsigned short f2bf(float f) {
    __hip_bfloat16 h = __float2bfloat16(f);
    return *reinterpret_cast<unsigned short*>(&h);
}
__device__ __forceinline__ float bf2f(unsigned short u) {
    __hip_bfloat16 h = *reinterpret_cast<__hip_bfloat16*>(&u);
    return __bfloat162float(h);
}
// Async global->LDS copy (CDNA5, ASYNCcnt-tracked). VDST VGPR holds the LDS
// byte offset (= low 32 bits of the generic pointer per flat-LDS aperture).
__device__ __forceinline__ void async_b128(void* lds, const void* gptr) {
    unsigned off = (unsigned)(unsigned long long)lds;
    asm volatile("global_load_async_to_lds_b128 %0, %1, off"
                 :: "v"(off), "v"(gptr) : "memory");
}
__device__ __forceinline__ void async_wait0() {
    asm volatile("s_wait_asynccnt 0" ::: "memory");
}

// ---------------------------------------------------------------------------
// Kernel 1: W f32 -> bf16, layout preserved (o,d). (o,d) IS the [n][k] layout
// the GEMM B-staging wants, so no transpose needed.
// ---------------------------------------------------------------------------
__global__ __launch_bounds__(256) void convert_w_bf16(
    const float* __restrict__ W, unsigned short* __restrict__ Wb)
{
    const size_t i = ((size_t)blockIdx.x * 256 + threadIdx.x) * 4;
    float4 f = *(const float4*)(W + i);
    uint2 p;
    p.x = (unsigned)f2bf(f.x) | ((unsigned)f2bf(f.y) << 16);
    p.y = (unsigned)f2bf(f.z) | ((unsigned)f2bf(f.w) << 16);
    *(uint2*)(Wb + i) = p;
}

// ---------------------------------------------------------------------------
// Kernel 2: projection GEMM  Y = X @ W^T + b, double-buffered K-loop with
// async B (and async A when A is bf16). Tile 32 tokens x 64 outputs (1 head),
// 8 waves x one 16x16 C tile, v_wmma_f32_16x16x32_bf16.
// Epilogues: Q -> RoPE+ELU, s-major bf16       Qf[b,h,s,d]
//            K -> RoPE+ELU, transposed bf16    Kt[b,h,d,s]
//            V -> transposed bf16              Vt[b,h,e,s]
//            O -> f32 final output (token, DMODEL)
// ---------------------------------------------------------------------------
enum { MODE_Q = 0, MODE_K = 1, MODE_V = 2, MODE_O = 3 };

template<int MODE, bool ABF16>
__global__ __launch_bounds__(256) void gemm_proj(
    const void* __restrict__ Xv, const unsigned short* __restrict__ Wb,
    const float* __restrict__ bias, void* __restrict__ Yv,
    const float* __restrict__ cosp, const float* __restrict__ sinp)
{
    __shared__ alignas(16) unsigned short As[2][32 * 32];   // [m][k]
    __shared__ alignas(16) unsigned short Bs[2][64 * 32];   // [n][k]
    __shared__ float Ct[32 * 65];                           // Q/K/V epilogue

    const int t    = threadIdx.x;
    const int lane = t & 31;
    const int w    = t >> 5;
    const int mi   = w & 1;
    const int ni   = w >> 1;
    const int mrow0 = blockIdx.x * 32;
    const int ncol0 = blockIdx.y * 64;

    // staging geometry
    const int am  = t >> 3,  ak  = (t & 7) * 4;   // f32 A path: 4 elems
    const int am2 = t >> 2,  ak2 = (t & 3) * 8;   // bf16 A path (t<128): 8
    const int bn  = t >> 2,  bk8 = (t & 3) * 8;   // B: 8 elems, contiguous k

    // fragment lane geometry (ISA 7.12.2)
    const int hiA  = (lane >= 16) ? 8  : 0;
    const int hiB  = (lane >= 16) ? 16 : 0;
    const int arow = mi * 16 + (lane & 15);
    const int bcol = ni * 16 + (lane & 15);

    const float* Xf = (const float*)Xv;
    const unsigned short* Xh = (const unsigned short*)Xv;
    const unsigned short* Bgp = Wb + (size_t)(ncol0 + bn) * DMODEL + bk8;

    v8f c = {};
    float4 areg;

    // ---- prologue: stage chunk 0 into buffer 0 ----
    if (ABF16) {
        if (t < 128)
            async_b128(&As[0][am2 * 32 + ak2],
                       Xh + (size_t)(mrow0 + am2) * DMODEL + ak2);
    } else {
        areg = *(const float4*)(Xf + (size_t)(mrow0 + am) * DMODEL + ak);
        uint2 p;
        p.x = (unsigned)f2bf(areg.x) | ((unsigned)f2bf(areg.y) << 16);
        p.y = (unsigned)f2bf(areg.z) | ((unsigned)f2bf(areg.w) << 16);
        *(uint2*)&As[0][am * 32 + ak] = p;
    }
    async_b128(&Bs[0][bn * 32 + bk8], Bgp);
    async_wait0();
    __syncthreads();

    for (int kc = 0; kc < DMODEL; kc += 32) {
        const int cur = (kc >> 5) & 1, nxt = cur ^ 1;
        const bool more = (kc + 32) < DMODEL;

        // ---- issue next chunk while computing current ----
        if (more) {
            if (ABF16) {
                if (t < 128)
                    async_b128(&As[nxt][am2 * 32 + ak2],
                               Xh + (size_t)(mrow0 + am2) * DMODEL + kc + 32 + ak2);
            } else {
                areg = *(const float4*)(Xf + (size_t)(mrow0 + am) * DMODEL + kc + 32 + ak);
            }
            async_b128(&Bs[nxt][bn * 32 + bk8], Bgp + kc + 32);
        }

        AFrag a, b;
        #pragma unroll
        for (int v = 0; v < 8; ++v) {
            const int kbA = ((v < 4) ? (2 * v) : (2 * v + 8)) + hiA;
            const int kbB = ((v < 4) ? (2 * v) : (2 * v + 8)) + hiA;
            a.u[v] = *(const unsigned int*)&As[cur][arow * 32 + kbA];
            b.u[v] = *(const unsigned int*)&Bs[cur][bcol * 32 + kbB];
        }
        c = __builtin_amdgcn_wmma_f32_16x16x32_bf16(
                false, a.v, false, b.v, (short)0, c, false, false);

        if (more && !ABF16) {
            uint2 p;
            p.x = (unsigned)f2bf(areg.x) | ((unsigned)f2bf(areg.y) << 16);
            p.y = (unsigned)f2bf(areg.z) | ((unsigned)f2bf(areg.w) << 16);
            *(uint2*)&As[nxt][am * 32 + ak] = p;
        }
        async_wait0();
        __syncthreads();
    }

    // bias (per output column)
    const float bv_ = bias[ncol0 + bcol];
    #pragma unroll
    for (int r = 0; r < 8; ++r) c[r] += bv_;

    const int h = blockIdx.y;   // 64-wide N tile == one head (Q/K/V)

    if constexpr (MODE == MODE_O) {
        float* Y = (float*)Yv;
        #pragma unroll
        for (int r = 0; r < 8; ++r) {
            const int m = mi * 16 + r + ((lane >> 4) << 3);
            Y[(size_t)(mrow0 + m) * DMODEL + ncol0 + bcol] = c[r];
        }
    } else {
        // Q/K/V share the LDS epilogue (RoPE needs cross-column pairs; K and
        // V are written transposed so stage-3 operands are contiguous).
        #pragma unroll
        for (int r = 0; r < 8; ++r) {
            const int m = mi * 16 + r + ((lane >> 4) << 3);
            Ct[m * 65 + bcol] = c[r];
        }
        __syncthreads();
        unsigned short* Y = (unsigned short*)Yv;
        #pragma unroll
        for (int it = 0; it < 8; ++it) {
            const int idx = t + it * 256;
            int m, d;
            if constexpr (MODE == MODE_Q) { m = idx >> 6; d = idx & 63; }
            else                          { m = idx & 31; d = idx >> 5; }
            const int g = mrow0 + m;
            const int bb = g >> 12, s = g & (SEQ - 1);
            float y = Ct[m * 65 + d];
            if constexpr (MODE != MODE_V) {
                const float xr  = Ct[m * 65 + (d ^ 32)];
                const float rot = (d < 32) ? -xr : xr;
                y = y * cosp[s * HDIM + d] + rot * sinp[s * HDIM + d];
                float f = (y > 0.f) ? (y + 1.f) : __expf(y);   // elu(y)+1
                y = fmaxf(f, 0.01f);
            }
            if constexpr (MODE == MODE_Q)
                Y[((size_t)(bb * NHEADS + h) * SEQ + s) * HDIM + d] = f2bf(y);
            else   // transposed: [b,h,d(or e),s]
                Y[((size_t)(bb * NHEADS + h) * HDIM + d) * SEQ + s] = f2bf(y);
        }
    }
}

// ---------------------------------------------------------------------------
// Kernel 3a: per (b,h,split)  partial KV = K^T V over an S-chunk of 512,
// plus partial K_sum. 512 WGs -> bandwidth-saturating. f32 partials.
// A = Kt[d][s], B = Vt[e][s]; both contiguous -> async double-buffered.
// ---------------------------------------------------------------------------
__global__ __launch_bounds__(256) void kv_partial_kernel(
    const unsigned short* __restrict__ Kt, const unsigned short* __restrict__ Vt,
    float* __restrict__ KVpart, float* __restrict__ Kspart)
{
    __shared__ alignas(16) unsigned short As[2][64 * 32];   // [d][s]
    __shared__ alignas(16) unsigned short Bs[2][64 * 32];   // [e][s]

    const int bh = blockIdx.x;
    const int sp = blockIdx.y;
    const int t = threadIdx.x, lane = t & 31, w = t >> 5;
    const int ti0 = w * 2, ti1 = w * 2 + 1;
    const int mi0 = ti0 & 3, ni0 = ti0 >> 2;
    const int mi1 = ti1 & 3, ni1 = ti1 >> 2;

    const int ad = t >> 2, as8 = (t & 3) * 8;   // 64 x 32, 8 elems/thread

    const int hiA = (lane >= 16) ? 8 : 0;
    const int hiB = (lane >= 16) ? 16 : 0;
    const int l15 = lane & 15;

    const unsigned short* Kg =
        Kt + (size_t)bh * HDIM * SEQ + (size_t)ad * SEQ + sp * SCHUNK + as8;
    const unsigned short* Vg =
        Vt + (size_t)bh * HDIM * SEQ + (size_t)ad * SEQ + sp * SCHUNK + as8;

    v8f c0 = {}, c1 = {};
    float ks0 = 0.f, ks1 = 0.f, ks2 = 0.f, ks3 = 0.f;

    async_b128(&As[0][ad * 32 + as8], Kg);
    async_b128(&Bs[0][ad * 32 + as8], Vg);
    async_wait0();
    __syncthreads();

    for (int sc = 0; sc < SCHUNK; sc += 32) {
        const int cur = (sc >> 5) & 1, nxt = cur ^ 1;
        const bool more = (sc + 32) < SCHUNK;
        if (more) {
            async_b128(&As[nxt][ad * 32 + as8], Kg + sc + 32);
            async_b128(&Bs[nxt][ad * 32 + as8], Vg + sc + 32);
        }

        if (t < 64) {   // K_sum partial: 4 independent accumulators for ILP
            #pragma unroll
            for (int k = 0; k < 32; k += 4) {
                ks0 += bf2f(As[cur][t * 32 + k + 0]);
                ks1 += bf2f(As[cur][t * 32 + k + 1]);
                ks2 += bf2f(As[cur][t * 32 + k + 2]);
                ks3 += bf2f(As[cur][t * 32 + k + 3]);
            }
        }

        AFrag a0, a1, b0, b1;
        #pragma unroll
        for (int v = 0; v < 8; ++v) {
            const int kbA = ((v < 4) ? (2 * v) : (2 * v + 8)) + hiA;
            const int kbB = ((v < 4) ? (2 * v) : (2 * v + 8)) + hiA;
            a0.u[v] = *(const unsigned int*)&As[cur][(mi0 * 16 + l15) * 32 + kbA];
            a1.u[v] = *(const unsigned int*)&As[cur][(mi1 * 16 + l15) * 32 + kbA];
            b0.u[v] = *(const unsigned int*)&Bs[cur][(ni0 * 16 + l15) * 32 + kbB];
            b1.u[v] = *(const unsigned int*)&Bs[cur][(ni1 * 16 + l15) * 32 + kbB];
        }
        c0 = __builtin_amdgcn_wmma_f32_16x16x32_bf16(
                 false, a0.v, false, b0.v, (short)0, c0, false, false);
        c1 = __builtin_amdgcn_wmma_f32_16x16x32_bf16(
                 false, a1.v, false, b1.v, (short)0, c1, false, false);

        async_wait0();
        __syncthreads();
    }

    // partial KV in [e][d] order (matches stage-4 B layout after reduction)
    float* dst = KVpart + ((size_t)bh * SPLIT + sp) * (HDIM * HDIM);
    #pragma unroll
    for (int r = 0; r < 8; ++r) {
        const int rr = r + ((lane >> 4) << 3);
        dst[(ni0 * 16 + l15) * HDIM + mi0 * 16 + rr] = c0[r];
        dst[(ni1 * 16 + l15) * HDIM + mi1 * 16 + rr] = c1[r];
    }
    if (t < 64)
        Kspart[((size_t)bh * SPLIT + sp) * HDIM + t] = (ks0 + ks1) + (ks2 + ks3);
}

// ---------------------------------------------------------------------------
// Kernel 3b: reduce the SPLIT partials -> KVt bf16 [e][d] and Ksum f32.
// Fixed summation order => deterministic.
// ---------------------------------------------------------------------------
__global__ __launch_bounds__(256) void kv_reduce_kernel(
    const float* __restrict__ KVpart, const float* __restrict__ Kspart,
    unsigned short* __restrict__ KVt, float* __restrict__ Ksum)
{
    const int bh = blockIdx.x;
    const int t = threadIdx.x;
    const float* src = KVpart + (size_t)bh * SPLIT * (HDIM * HDIM);
    #pragma unroll
    for (int i = 0; i < (HDIM * HDIM) / 256; ++i) {
        const int idx = t + i * 256;
        float acc = 0.f;
        #pragma unroll
        for (int p = 0; p < SPLIT; ++p) acc += src[p * (HDIM * HDIM) + idx];
        KVt[(size_t)bh * HDIM * HDIM + idx] = f2bf(acc);
    }
    if (t < HDIM) {
        const float* ss = Kspart + (size_t)bh * SPLIT * HDIM;
        float acc = 0.f;
        #pragma unroll
        for (int p = 0; p < SPLIT; ++p) acc += ss[p * HDIM + t];
        Ksum[bh * HDIM + t] = acc;
    }
}

// ---------------------------------------------------------------------------
// Kernel 4: out = (Qf @ KV) / (clip(Qf . K_sum, 1) + 1e-6)
// Tile 32 tokens x 64 outputs (one head), K = 64 (two WMMA chunks).
// ---------------------------------------------------------------------------
__global__ __launch_bounds__(256) void attn_kernel(
    const unsigned short* __restrict__ Qf, const unsigned short* __restrict__ KVt,
    const float* __restrict__ Ksum, unsigned short* __restrict__ Att)
{
    __shared__ alignas(16) unsigned short Qs[32 * 64];    // [m][d]
    __shared__ alignas(16) unsigned short KVs[64 * 64];   // [e][d]
    __shared__ float KsS[64];
    __shared__ float Ns[32];

    const int bh = blockIdx.y;
    const int b = bh >> 4, h = bh & 15;
    const int s0 = blockIdx.x * 32;
    const int t = threadIdx.x, lane = t & 31, w = t >> 5;
    const int mi = w & 1, ni = w >> 1;

    {   // stage Qf (32x64) and KVt (64x64) asynchronously
        const int m = t >> 3, d8 = (t & 7) * 8;
        async_b128(&Qs[m * 64 + d8],
                   Qf + ((size_t)bh * SEQ + s0 + m) * HDIM + d8);
        const int e = t >> 2, d16 = (t & 3) * 16;
        const unsigned short* src = KVt + (size_t)bh * HDIM * HDIM + e * HDIM + d16;
        async_b128(&KVs[e * 64 + d16], src);
        async_b128(&KVs[e * 64 + d16 + 8], src + 8);
    }
    if (t < 64) KsS[t] = Ksum[bh * HDIM + t];
    async_wait0();
    __syncthreads();

    if (t < 32) {
        float a0 = 0.f, a1 = 0.f, a2 = 0.f, a3 = 0.f;
        #pragma unroll
        for (int d = 0; d < HDIM; d += 4) {
            a0 += bf2f(Qs[t * 64 + d + 0]) * KsS[d + 0];
            a1 += bf2f(Qs[t * 64 + d + 1]) * KsS[d + 1];
            a2 += bf2f(Qs[t * 64 + d + 2]) * KsS[d + 2];
            a3 += bf2f(Qs[t * 64 + d + 3]) * KsS[d + 3];
        }
        Ns[t] = fmaxf((a0 + a1) + (a2 + a3), 1.0f) + 1e-6f;
    }

    const int hiA = (lane >= 16) ? 8 : 0;
    const int hiB = (lane >= 16) ? 16 : 0;
    const int arow = mi * 16 + (lane & 15);
    const int bcol = ni * 16 + (lane & 15);

    v8f c = {};
    #pragma unroll
    for (int kc = 0; kc < HDIM; kc += 32) {
        AFrag a, bfr;
        #pragma unroll
        for (int v = 0; v < 8; ++v) {
            const int kbA = ((v < 4) ? (2 * v) : (2 * v + 8)) + hiA;
            const int kbB = ((v < 4) ? (2 * v) : (2 * v + 8)) + hiA;
            a.u[v]   = *(const unsigned int*)&Qs[arow * 64 + kc + kbA];
            bfr.u[v] = *(const unsigned int*)&KVs[bcol * 64 + kc + kbB];
        }
        c = __builtin_amdgcn_wmma_f32_16x16x32_bf16(
                false, a.v, false, bfr.v, (short)0, c, false, false);
    }
    __syncthreads();   // Ns visible

    #pragma unroll
    for (int r = 0; r < 8; ++r) {
        const int m = mi * 16 + r + ((lane >> 4) << 3);
        const float val = c[r] / Ns[m];
        Att[(size_t)(b * SEQ + s0 + m) * DMODEL + h * HDIM + bcol] = f2bf(val);
    }
}

// ---------------------------------------------------------------------------
// Launch
// ---------------------------------------------------------------------------
extern "C" void kernel_launch(void* const* d_in, const int* in_sizes, int n_in,
                              void* d_out, int out_size, void* d_ws, size_t ws_size,
                              hipStream_t stream) {
    const float* query = (const float*)d_in[0];
    const float* key_  = (const float*)d_in[1];
    const float* value = (const float*)d_in[2];
    const float* cosp  = (const float*)d_in[3];
    const float* sinp  = (const float*)d_in[4];
    const float* Wq = (const float*)d_in[5];  const float* bq = (const float*)d_in[6];
    const float* Wk = (const float*)d_in[7];  const float* bk = (const float*)d_in[8];
    const float* Wv = (const float*)d_in[9];  const float* bv = (const float*)d_in[10];
    const float* Wo = (const float*)d_in[11]; const float* bo = (const float*)d_in[12];
    float* out = (float*)d_out;

    char* ws = (char*)d_ws;
    size_t off = 0;
    auto alloc = [&](size_t bytes) -> void* {
        void* p = ws + off;
        off = (off + bytes + 255) & ~(size_t)255;
        return p;
    };
    const size_t WB  = (size_t)DMODEL * DMODEL * 2;     // 2 MB
    const size_t ACT = (size_t)NTOK * DMODEL * 2;       // 32 MB
    unsigned short* Wbq = (unsigned short*)alloc(WB);
    unsigned short* Wbk = (unsigned short*)alloc(WB);
    unsigned short* Wbv = (unsigned short*)alloc(WB);
    unsigned short* Wbo = (unsigned short*)alloc(WB);
    unsigned short* QfB = (unsigned short*)alloc(ACT);
    unsigned short* KtB = (unsigned short*)alloc(ACT);
    unsigned short* VtB = (unsigned short*)alloc(ACT);
    unsigned short* AttB = (unsigned short*)alloc(ACT);
    unsigned short* KVt = (unsigned short*)alloc((size_t)BATCH * NHEADS * HDIM * HDIM * 2);
    float* KVpart = (float*)alloc((size_t)BATCH * NHEADS * SPLIT * HDIM * HDIM * 4);
    float* Kspart = (float*)alloc((size_t)BATCH * NHEADS * SPLIT * HDIM * 4);
    float* KsumB  = (float*)alloc((size_t)BATCH * NHEADS * HDIM * 4);
    if (off > ws_size) return;   // workspace too small: bail deterministically

    convert_w_bf16<<<DMODEL * DMODEL / 1024, 256, 0, stream>>>(Wq, Wbq);
    convert_w_bf16<<<DMODEL * DMODEL / 1024, 256, 0, stream>>>(Wk, Wbk);
    convert_w_bf16<<<DMODEL * DMODEL / 1024, 256, 0, stream>>>(Wv, Wbv);
    convert_w_bf16<<<DMODEL * DMODEL / 1024, 256, 0, stream>>>(Wo, Wbo);

    dim3 gg(NTOK / 32, DMODEL / 64);   // (512, 16)
    gemm_proj<MODE_Q, false><<<gg, 256, 0, stream>>>(query, Wbq, bq, QfB, cosp, sinp);
    gemm_proj<MODE_K, false><<<gg, 256, 0, stream>>>(key_,  Wbk, bk, KtB, cosp, sinp);
    gemm_proj<MODE_V, false><<<gg, 256, 0, stream>>>(value, Wbv, bv, VtB, nullptr, nullptr);

    kv_partial_kernel<<<dim3(BATCH * NHEADS, SPLIT), 256, 0, stream>>>(
        KtB, VtB, KVpart, Kspart);
    kv_reduce_kernel<<<BATCH * NHEADS, 256, 0, stream>>>(
        KVpart, Kspart, KVt, KsumB);
    attn_kernel<<<dim3(SEQ / 32, BATCH * NHEADS), 256, 0, stream>>>(
        QfB, KVt, KsumB, AttB);

    gemm_proj<MODE_O, true><<<gg, 256, 0, stream>>>(AttB, Wbo, bo, out, nullptr, nullptr);
}
